// Model_3599182594396
// MI455X (gfx1250) — compile-verified, matrix operand order checked by reading, not run on hardware
//
#include <hip/hip_runtime.h>

// ---------------------------------------------------------------------------
// GraphSAGE (4 SAGEConv mean-aggr layers) + global mean pool + MLP head
// for MI455X / gfx1250.
//  - GEMMs: v_wmma_f32_16x16x32_bf16, f32 accumulate, weights staged to LDS
//    by the Tensor Data Mover (tensor_load_to_lds with D# row padding).
//  - Wave tile 32x64 (2 row banks x 4 col tiles): 2 WMMAs per B fragment,
//    ~110 VGPRs -> no spills.
//  - Edge aggregation: f32 global atomics (memory-bound, ~200MB/layer).
// ---------------------------------------------------------------------------

#define N_NODES   200000
#define N_EDGES   400000
#define N_GRAPHS  4096
#define IN_DIM    26
#define HID       128
#define CLS_HID   256
#define N_LAYERS  3
#define NUM_CLASSES 2

#define USE_TDM 1

typedef __attribute__((ext_vector_type(16))) __bf16 v16bf;
typedef __attribute__((ext_vector_type(8)))  float  v8f;
typedef __attribute__((ext_vector_type(4)))  unsigned int u32x4;
typedef __attribute__((ext_vector_type(8)))  unsigned int u32x8;

// ---------------- fragment loaders ----------------
// A-fragment (16x32 bf16, ISA layout): lane l<16 -> row M=l, K = {0..7,16..23};
// lane l>=16 -> row M=l-16, K = {8..15,24..31}.  ho = (lane>>4)*8.
__device__ __forceinline__ v16bf load_frag_bf16(const __bf16* base, int ho) {
    union { uint4 q[2]; v16bf v; } u;
    u.q[0] = *(const uint4*)(base + ho);        // K = ho..ho+7
    u.q[1] = *(const uint4*)(base + 16 + ho);   // K = 16+ho..23+ho
    return u.v;
}

__device__ __forceinline__ v16bf load_frag_f32_scaled(const float* base, int ho, float s) {
    float4 a0 = *(const float4*)(base + ho);
    float4 a1 = *(const float4*)(base + ho + 4);
    float4 a2 = *(const float4*)(base + 16 + ho);
    float4 a3 = *(const float4*)(base + 16 + ho + 4);
    v16bf v;
    v[0]=(__bf16)(a0.x*s); v[1]=(__bf16)(a0.y*s); v[2]=(__bf16)(a0.z*s); v[3]=(__bf16)(a0.w*s);
    v[4]=(__bf16)(a1.x*s); v[5]=(__bf16)(a1.y*s); v[6]=(__bf16)(a1.z*s); v[7]=(__bf16)(a1.w*s);
    v[8]=(__bf16)(a2.x*s); v[9]=(__bf16)(a2.y*s); v[10]=(__bf16)(a2.z*s); v[11]=(__bf16)(a2.w*s);
    v[12]=(__bf16)(a3.x*s); v[13]=(__bf16)(a3.y*s); v[14]=(__bf16)(a3.z*s); v[15]=(__bf16)(a3.w*s);
    return v;
}

__device__ __forceinline__ v16bf load_frag_lds(const char* p) {
    union { uint4 q[2]; v16bf v; } u;
    u.q[0] = *(const uint4*)p;         // K = ho..ho+7   (2 bytes/elem)
    u.q[1] = *(const uint4*)(p + 32);  // K = 16+ho..
    return u.v;
}

// ---------------- small kernels ----------------
__global__ void k_zero(float* __restrict__ p, size_t n) {
    size_t t = ((size_t)blockIdx.x * 256 + threadIdx.x) * 4;
    #pragma unroll
    for (int j = 0; j < 4; ++j)
        if (t + j < n) p[t + j] = 0.f;
}

__global__ void k_deg(const int* __restrict__ dst, float* __restrict__ deg) {
    size_t t = (size_t)blockIdx.x * 256 + threadIdx.x;
    if (t < N_EDGES) atomicAdd(&deg[dst[t]], 1.0f);
}

__global__ void k_inv(float* __restrict__ inv) {
    size_t t = (size_t)blockIdx.x * 256 + threadIdx.x;
    if (t < N_NODES) inv[t] = 1.0f / fmaxf(inv[t], 1.0f);
}

__global__ void k_scatter26(const float* __restrict__ x, const int* __restrict__ src,
                            const int* __restrict__ dst, float* __restrict__ aggX) {
    size_t t = (size_t)blockIdx.x * 256 + threadIdx.x;
    if (t >= (size_t)N_EDGES * 32) return;
    int e = (int)(t >> 5), f = (int)(t & 31);
    if (f < IN_DIM)
        atomicAdd(&aggX[(size_t)dst[e] * IN_DIM + f], x[(size_t)src[e] * IN_DIM + f]);
}

// 4 features per thread: one b64 load of 4 bf16, 4 f32 atomics
__global__ void k_scatter128(const __bf16* __restrict__ h, const int* __restrict__ src,
                             const int* __restrict__ dst, float* __restrict__ agg) {
    size_t t = (size_t)blockIdx.x * 256 + threadIdx.x;
    if (t >= (size_t)N_EDGES * 32) return;
    int e = (int)(t >> 5), f4 = (int)(t & 31) * 4;
    int s = src[e], d = dst[e];
    union { unsigned long long u; __bf16 b[4]; } v;
    v.u = *(const unsigned long long*)(h + (size_t)s * HID + f4);
    float* ap = agg + (size_t)d * HID + f4;
    atomicAdd(ap + 0, (float)v.b[0]);
    atomicAdd(ap + 1, (float)v.b[1]);
    atomicAdd(ap + 2, (float)v.b[2]);
    atomicAdd(ap + 3, (float)v.b[3]);
}

// A1[n][0..63] = [ aggX*inv (26) | 0 pad | x (26) | 0 pad ]  in bf16
__global__ void k_build_A1(const float* __restrict__ x, const float* __restrict__ aggX,
                           const float* __restrict__ inv, __bf16* __restrict__ A1) {
    size_t t = (size_t)blockIdx.x * 256 + threadIdx.x;
    if (t >= (size_t)N_NODES * 64) return;
    int n = (int)(t >> 6), k = (int)(t & 63);
    float v = 0.f;
    if (k < IN_DIM)                  v = aggX[(size_t)n * IN_DIM + k] * inv[n];
    else if (k >= 32 && k < 32 + IN_DIM) v = x[(size_t)n * IN_DIM + (k - 32)];
    A1[t] = (__bf16)v;
}

// fuse weights to bf16, N x K row-major (row = output channel)
__global__ void k_prep(const float* __restrict__ W1l, const float* __restrict__ W1r,
                       const float* __restrict__ Wl,  const float* __restrict__ Wr,
                       const float* __restrict__ Wlin1,
                       __bf16* __restrict__ WB1, __bf16* __restrict__ WB,
                       __bf16* __restrict__ WC1) {
    int t = blockIdx.x * 256 + threadIdx.x;
    if (t < 128 * 64) {                       // WB1: [W1l | pad | W1r | pad], K=64
        int n = t >> 6, k = t & 63;
        float v = 0.f;
        if (k < IN_DIM)                    v = W1l[n * IN_DIM + k];
        else if (k >= 32 && k < 32 + IN_DIM) v = W1r[n * IN_DIM + (k - 32)];
        WB1[t] = (__bf16)v;
    } else if (t < 128 * 64 + N_LAYERS * 128 * 256) {  // WB[i]: [Wl_i | Wr_i], K=256
        int u = t - 128 * 64;
        int layer = u / (128 * 256), r = u % (128 * 256);
        int n = r >> 8, k = r & 255;
        float v = (k < HID) ? Wl[((size_t)layer * HID + n) * HID + k]
                            : Wr[((size_t)layer * HID + n) * HID + (k - HID)];
        WB[u] = (__bf16)v;
    } else if (t < 128 * 64 + N_LAYERS * 128 * 256 + CLS_HID * HID) {
        int u = t - (128 * 64 + N_LAYERS * 128 * 256);
        WC1[u] = (__bf16)Wlin1[u];            // already (256,128) = N x K
    }
}

// ---------------- WMMA GEMM ----------------
// Block = 256 threads = 8 waves; block tile 128 rows x 128 cols.
// Wave tile 32 rows x 64 cols: wave w -> row-tile (w&3), column half (w>>2).
// Two 16-row banks share each B fragment -> 2 WMMAs per ds_load pair, with
// acc = 64 VGPRs (no spills).
// B (N x K bf16) staged in LDS by TDM with 16B row padding (conflict-free).
// AMODE 0: A = Abf, M x K bf16 row-major.
// AMODE 1: SAGE, K=256: k<128 from agg f32 (scaled by inv[m]), k>=128 from h bf16.
// OUTF32RELU: store relu(acc+bias) to f32, else bf16.
template <int K, int AMODE, int OUTF32RELU>
__global__ void k_gemm(const __bf16* __restrict__ Abf, const float* __restrict__ Aagg,
                       const __bf16* __restrict__ Ah,  const float* __restrict__ inv,
                       const __bf16* __restrict__ Bw,  const float* __restrict__ bias,
                       __bf16* __restrict__ Obf, float* __restrict__ Of,
                       int M, int ldO) {
    extern __shared__ char smem[];
    const int ROWB = K * 2 + 16;
    const int tid  = threadIdx.x;
    const int cofs = blockIdx.y * 128;

#if USE_TDM
    // Stage 128 rows x K bf16 of B via the Tensor Data Mover. 2D tile in dword
    // units; D#.pad adds 4 dwords (16B) after every K/2-dword row -> LDS row
    // stride = ROWB. Tensor ops ignore EXEC; one wave issues, waits TENSORcnt.
    if (tid == 0) {
        unsigned long long ga = (unsigned long long)(Bw + (size_t)cofs * K);
        unsigned lds = (unsigned)(unsigned long long)smem;   // LDS byte offset
        const unsigned ROWDW = K / 2;                        // dwords per row
        const unsigned PADI  = (K == 256) ? 6u : (K == 128) ? 5u : 4u; // 2^(c+1)
        u32x4 g0;
        g0[0] = 1u;                                          // count=1, load-D#
        g0[1] = lds;                                         // lds_addr
        g0[2] = (unsigned)(ga & 0xffffffffu);                // global_addr[31:0]
        g0[3] = (unsigned)((ga >> 32) & 0x1ffffffu) | (2u << 30); // [56:32]|type=2
        u32x8 g1;
        g1[0] = (2u << 16) | (1u << 20) | (PADI << 22) | (3u << 25); // 4B elems,
                                                  // pad_enable, interval, 4 dw pad
        g1[1] = (ROWDW & 0xffffu) << 16;          // tensor_dim0[15:0]
        g1[2] = (ROWDW >> 16) | (128u << 16);     // tensor_dim0[31:16]|dim1[15:0]
        g1[3] = (ROWDW << 16);                    // tile_dim0
        g1[4] = 128u;                             // tile_dim1 (tile_dim2=0)
        g1[5] = ROWDW;                            // tensor_dim0_stride[31:0]
        g1[6] = 0u;
        g1[7] = 0u;
        asm volatile("tensor_load_to_lds %0, %1" :: "s"(g0), "s"(g1) : "memory");
        __builtin_amdgcn_s_wait_tensorcnt(0);
    }
#else
    for (int i = tid; i < 128 * (K / 8); i += 256) {
        int n = i / (K / 8), kk = (i % (K / 8)) * 8;
        *(uint4*)(smem + n * ROWB + kk * 2) =
            *(const uint4*)(Bw + (size_t)(cofs + n) * K + kk);
    }
#endif
    __syncthreads();

    const int wave  = tid >> 5, lane = tid & 31;
    const int tile  = blockIdx.x * 4 + (wave & 3);   // 32-row tiles
    const int chalf = wave >> 2;                     // column half (0/1) -> 64 cols
    if (tile * 32 >= M) return;                 // wave-uniform: EXEC stays all-1
    const int m0   = tile * 32;
    const int mr0  = m0 + (lane & 15);
    const int mr1  = mr0 + 16;
    const int half = lane >> 4;
    const int ho   = half * 8;
    const int nbase = chalf * 64;               // local column base in LDS tile

    v8f acc0[4], acc1[4];
    #pragma unroll
    for (int n = 0; n < 4; ++n) {
        acc0[n] = (v8f){0.f,0.f,0.f,0.f,0.f,0.f,0.f,0.f};
        acc1[n] = (v8f){0.f,0.f,0.f,0.f,0.f,0.f,0.f,0.f};
    }
    const float s0 = (AMODE == 1) ? inv[mr0] : 0.f;
    const float s1 = (AMODE == 1) ? inv[mr1] : 0.f;

    #pragma unroll
    for (int kt = 0; kt < K / 32; ++kt) {
        v16bf a0, a1;
        if (AMODE == 0) {
            a0 = load_frag_bf16(Abf + (size_t)mr0 * K + kt * 32, ho);
            a1 = load_frag_bf16(Abf + (size_t)mr1 * K + kt * 32, ho);
        } else {
            if (kt < 4) {
                a0 = load_frag_f32_scaled(Aagg + (size_t)mr0 * HID + kt * 32, ho, s0);
                a1 = load_frag_f32_scaled(Aagg + (size_t)mr1 * HID + kt * 32, ho, s1);
            } else {
                a0 = load_frag_bf16(Ah + (size_t)mr0 * HID + (kt - 4) * 32, ho);
                a1 = load_frag_bf16(Ah + (size_t)mr1 * HID + (kt - 4) * 32, ho);
            }
        }
        #pragma unroll
        for (int nt = 0; nt < 4; ++nt) {
            const char* bp = smem + (nbase + nt * 16 + (lane & 15)) * ROWB
                           + (kt * 32 + ho) * 2;
            v16bf b = load_frag_lds(bp);
            acc0[nt] = __builtin_amdgcn_wmma_f32_16x16x32_bf16(
                false, a0, false, b, (short)0, acc0[nt], false, false);
            acc1[nt] = __builtin_amdgcn_wmma_f32_16x16x32_bf16(
                false, a1, false, b, (short)0, acc1[nt], false, false);
        }
    }

    // C/D layout: VGPR r, lane l<16 -> (M=r, N=l); lane>=16 -> (M=8+r, N=l-16)
    #pragma unroll
    for (int nt = 0; nt < 4; ++nt) {
        int col = cofs + nbase + nt * 16 + (lane & 15);
        float bv = bias[col];
        #pragma unroll
        for (int r = 0; r < 8; ++r) {
            int row0 = m0 + r + half * 8;
            int row1 = row0 + 16;
            float v0 = acc0[nt][r] + bv;
            float v1 = acc1[nt][r] + bv;
            if (OUTF32RELU) {
                Of[(size_t)row0 * ldO + col] = v0 > 0.f ? v0 : 0.f;
                Of[(size_t)row1 * ldO + col] = v1 > 0.f ? v1 : 0.f;
            } else {
                Obf[(size_t)row0 * ldO + col] = (__bf16)v0;
                Obf[(size_t)row1 * ldO + col] = (__bf16)v1;
            }
        }
    }
}

// ---------------- pooling + head ----------------
__global__ void k_pool(const __bf16* __restrict__ h, const int* __restrict__ batch,
                       float* __restrict__ gsum) {
    size_t t = (size_t)blockIdx.x * 256 + threadIdx.x;
    if (t >= (size_t)N_NODES * 32) return;
    int n = (int)(t >> 5), f4 = (int)(t & 31) * 4;
    int g = batch[n];
    union { unsigned long long u; __bf16 b[4]; } v;
    v.u = *(const unsigned long long*)(h + (size_t)n * HID + f4);
    float* gp = gsum + (size_t)g * HID + f4;
    atomicAdd(gp + 0, (float)v.b[0]);
    atomicAdd(gp + 1, (float)v.b[1]);
    atomicAdd(gp + 2, (float)v.b[2]);
    atomicAdd(gp + 3, (float)v.b[3]);
}

__global__ void k_cnt(const int* __restrict__ batch, float* __restrict__ gcnt) {
    size_t t = (size_t)blockIdx.x * 256 + threadIdx.x;
    if (t < N_NODES) atomicAdd(&gcnt[batch[t]], 1.0f);
}

__global__ void k_gdiv(const float* __restrict__ gsum, const float* __restrict__ gcnt,
                       __bf16* __restrict__ gbf) {
    size_t t = (size_t)blockIdx.x * 256 + threadIdx.x;
    if (t >= (size_t)N_GRAPHS * HID) return;
    int g = (int)(t >> 7);
    gbf[t] = (__bf16)(gsum[t] / fmaxf(gcnt[g], 1.0f));
}

__global__ void k_lin2(const float* __restrict__ gh, const float* __restrict__ W2,
                       const float* __restrict__ b2, float* __restrict__ out) {
    int t = blockIdx.x * 256 + threadIdx.x;
    if (t >= N_GRAPHS * NUM_CLASSES) return;
    int g = t >> 1, c = t & 1;
    const float* a = gh + (size_t)g * CLS_HID;
    const float* w = W2 + (size_t)c * CLS_HID;
    float s = b2[c];
    #pragma unroll 8
    for (int k = 0; k < CLS_HID; ++k) s += a[k] * w[k];
    out[t] = s;
}

// ---------------- host launch ----------------
static inline void zero_f32(float* p, size_t n, hipStream_t s) {
    k_zero<<<dim3((unsigned)((n + 1023) / 1024)), 256, 0, s>>>(p, n);
}

extern "C" void kernel_launch(void* const* d_in, const int* in_sizes, int n_in,
                              void* d_out, int out_size, void* d_ws, size_t ws_size,
                              hipStream_t stream) {
    const float* x      = (const float*)d_in[0];
    const int*   src    = (const int*)d_in[1];
    const int*   dst    = src + N_EDGES;
    const int*   batch  = (const int*)d_in[2];
    const float* W1l    = (const float*)d_in[3];
    const float* b1l    = (const float*)d_in[4];
    const float* W1r    = (const float*)d_in[5];
    const float* Wl     = (const float*)d_in[6];
    const float* bl     = (const float*)d_in[7];
    const float* Wr     = (const float*)d_in[8];
    const float* Wlin1  = (const float*)d_in[9];
    const float* blin1  = (const float*)d_in[10];
    const float* Wlin2  = (const float*)d_in[11];
    const float* blin2  = (const float*)d_in[12];
    float* out = (float*)d_out;

    char* w = (char*)d_ws;
    size_t o = 0;
    auto alloc = [&](size_t bytes) { void* p = w + o; o = (o + bytes + 255) & ~(size_t)255; return p; };

    float*  inv  = (float*)alloc((size_t)N_NODES * 4);
    float*  agg  = (float*)alloc((size_t)N_NODES * HID * 4);   // 102.4 MB, reused
    __bf16* hA   = (__bf16*)alloc((size_t)N_NODES * HID * 2);
    __bf16* hB   = (__bf16*)alloc((size_t)N_NODES * HID * 2);
    __bf16* WB1  = (__bf16*)alloc((size_t)128 * 64 * 2);
    __bf16* WB   = (__bf16*)alloc((size_t)N_LAYERS * 128 * 256 * 2);
    __bf16* WC1  = (__bf16*)alloc((size_t)CLS_HID * HID * 2);
    float*  gsum = (float*)alloc((size_t)N_GRAPHS * HID * 4);
    float*  gcnt = (float*)alloc((size_t)N_GRAPHS * 4);
    __bf16* gbf  = (__bf16*)alloc((size_t)N_GRAPHS * HID * 2);
    float*  gh   = (float*)alloc((size_t)N_GRAPHS * CLS_HID * 4);

    // layer-1 staging overlaid on (later-lived) agg region
    float*  aggX = agg;                                   // N_NODES x 26 f32
    __bf16* A1   = (__bf16*)((char*)agg + (32u << 20));   // N_NODES x 64 bf16

    // degree -> inv_cnt
    zero_f32(inv, N_NODES, stream);
    k_deg<<<dim3((N_EDGES + 255) / 256), 256, 0, stream>>>(dst, inv);
    k_inv<<<dim3((N_NODES + 255) / 256), 256, 0, stream>>>(inv);

    // layer 1: scatter x, build padded bf16 A, fused weights, WMMA GEMM
    zero_f32(aggX, (size_t)N_NODES * IN_DIM, stream);
    k_scatter26<<<dim3((unsigned)(((size_t)N_EDGES * 32 + 255) / 256)), 256, 0, stream>>>(
        x, src, dst, aggX);
    k_build_A1<<<dim3((unsigned)(((size_t)N_NODES * 64 + 255) / 256)), 256, 0, stream>>>(x, aggX, inv, A1);
    k_prep<<<dim3((128 * 64 + N_LAYERS * 128 * 256 + CLS_HID * HID + 255) / 256), 256, 0, stream>>>(
        W1l, W1r, Wl, Wr, Wlin1, WB1, WB, WC1);

    const unsigned GB = (unsigned)((N_NODES / 32 + 3) / 4);  // 1563 blocks (4 row-tiles each)
    k_gemm<64, 0, 0><<<dim3(GB, 1), 256, 128 * (64 * 2 + 16), stream>>>(
        A1, nullptr, nullptr, nullptr, WB1, b1l, hA, nullptr, N_NODES, HID);

    // layers 2..4: f32 atomic scatter + fused SAGE GEMM (agg*inv ++ h)
    __bf16* hcur = hA; __bf16* hnext = hB;
    for (int i = 0; i < N_LAYERS; ++i) {
        zero_f32(agg, (size_t)N_NODES * HID, stream);
        k_scatter128<<<dim3((unsigned)(((size_t)N_EDGES * 32 + 255) / 256)), 256, 0, stream>>>(
            hcur, src, dst, agg);
        k_gemm<256, 1, 0><<<dim3(GB, 1), 256, 128 * (256 * 2 + 16), stream>>>(
            nullptr, agg, hcur, inv, WB + (size_t)i * 128 * 256, bl + (size_t)i * HID,
            hnext, nullptr, N_NODES, HID);
        __bf16* t = hcur; hcur = hnext; hnext = t;
    }

    // global mean pool
    zero_f32(gsum, (size_t)N_GRAPHS * HID, stream);
    zero_f32(gcnt, (size_t)N_GRAPHS, stream);
    k_pool<<<dim3((unsigned)(((size_t)N_NODES * 32 + 255) / 256)), 256, 0, stream>>>(hcur, batch, gsum);
    k_cnt<<<dim3((N_NODES + 255) / 256), 256, 0, stream>>>(batch, gcnt);
    k_gdiv<<<dim3((N_GRAPHS * HID + 255) / 256), 256, 0, stream>>>(gsum, gcnt, gbf);

    // lin1 (4096x128 @ 128x256, relu) via WMMA; lin2 tiny VALU
    k_gemm<128, 0, 1><<<dim3(N_GRAPHS / 128, 2), 256, 128 * (128 * 2 + 16), stream>>>(
        gbf, nullptr, nullptr, nullptr, WC1, blin1, nullptr, gh, N_GRAPHS, CLS_HID);
    k_lin2<<<dim3((N_GRAPHS * NUM_CLASSES + 255) / 256), 256, 0, stream>>>(gh, Wlin2, blin2, out);
}
